// HeteroEncoderDecoderModel_43885975830667
// MI455X (gfx1250) — compile-verified, matrix operand order checked by reading, not run on hardware
//
#include <hip/hip_runtime.h>
#include <hip/hip_bf16.h>
#include <math.h>

// ---------------- model dims ----------------
#define N1_ 8192
#define N2_ 16384
#define E11_ 16384
#define E22_ 32768
#define E12_ 16384
#define T_HIST_ 6
#define T_FUT_ 12
#define H_DIM_ 192
#define MSG_ 128
#define HID_ 256
#define DMOD_ 256
#define NHEAD_ 8
#define DH_ 32
#define FFN_ 512
#define DCH_ 1024   // decoder node chunk

#define CDIV(a,b) (((a)+(b)-1)/(b))

typedef __attribute__((ext_vector_type(16))) _Float16 v16h;
typedef __attribute__((ext_vector_type(8)))  float    v8f;

__device__ __forceinline__ float sig_(float x){ return 1.f/(1.f+__expf(-x)); }
__device__ __forceinline__ float softplus_(float x){ return x>0.f ? x+log1pf(__expf(-x)) : log1pf(__expf(x)); }

// ---------------- WMMA GEMM: C = act(A[MxK] + bias) (+C if ACCUM) ----------------
// One wave computes a 64x16 strip (4 accumulators): B k-slab loaded once per K-step,
// reused by 4 back-to-back v_wmma_f32_16x16x32_f16.
// Fast path (wave-uniform interior, K%32==0): unconditional loads, A via float4
// (provably 16B-aligned: workspace buffers are 256B aligned and K%32==0).
// Boundary path: 32-bit clamped offsets + v_cndmask (branch-free, no saveexec).
template<int ACT, int ACCUM>
__global__ __launch_bounds__(128)
void gemm_k(const float* __restrict__ A, const float* __restrict__ W,
            const float* __restrict__ bias, float* __restrict__ C,
            int M, int K, int N) {
  const int lane = threadIdx.x & 31;
  const int wave = threadIdx.x >> 5;
  const int tilesN = (N + 15) >> 4;
  const int tilesM = (M + 63) >> 6;
  const int numTiles = tilesM * tilesN;
  const int tile = blockIdx.x * 4 + wave;
  if (tile >= numTiles) return;
  const int tm = tile / tilesN;
  const int tn = tile % tilesN;
  const int hi = lane >> 4;    // half-wave selects K sub-block
  const int ml = lane & 15;    // A row-in-subtile / B+D column
  const int rowBase = tm * 64;
  const int col = tn * 16 + ml;
  const bool fullM = (rowBase + 64 <= M);
  const bool fullN = (tn * 16 + 16 <= N);
  const bool fullK = ((K & 31) == 0);
  const bool colOK = col < N;

  v8f acc[4];
#pragma unroll
  for (int s = 0; s < 4; ++s) acc[s] = (v8f){0.f,0.f,0.f,0.f,0.f,0.f,0.f,0.f};

  if (fullM && fullN && fullK) {
    // -------- fast path --------
    for (int k0 = 0; k0 < K; k0 += 32) {
      float fb[16];
#pragma unroll
      for (int j = 0; j < 16; ++j) {
        const int k = k0 + ((j >> 3) << 4) + (hi << 3) + (j & 7);
        fb[j] = W[(unsigned)(k * N + col)];
      }
      v16h b;
#pragma unroll
      for (int j = 0; j < 16; ++j) b[j] = (_Float16)fb[j];
#pragma unroll
      for (int s = 0; s < 4; ++s) {
        const float4* Ap = reinterpret_cast<const float4*>(
            A + (long)(rowBase + s * 16 + ml) * K + k0 + (hi << 3));
        const float4 a0 = Ap[0];   // k offsets 0..3
        const float4 a1 = Ap[1];   // 4..7
        const float4 a2 = Ap[4];   // 16..19
        const float4 a3 = Ap[5];   // 20..23
        v16h a;
        a[0] = (_Float16)a0.x; a[1] = (_Float16)a0.y; a[2] = (_Float16)a0.z; a[3] = (_Float16)a0.w;
        a[4] = (_Float16)a1.x; a[5] = (_Float16)a1.y; a[6] = (_Float16)a1.z; a[7] = (_Float16)a1.w;
        a[8] = (_Float16)a2.x; a[9] = (_Float16)a2.y; a[10] = (_Float16)a2.z; a[11] = (_Float16)a2.w;
        a[12] = (_Float16)a3.x; a[13] = (_Float16)a3.y; a[14] = (_Float16)a3.z; a[15] = (_Float16)a3.w;
        acc[s] = __builtin_amdgcn_wmma_f32_16x16x32_f16(false, a, false, b, (short)0, acc[s], false, false);
      }
    }
  } else {
    // -------- boundary path: 32-bit clamped offsets + selects --------
    for (int k0 = 0; k0 < K; k0 += 32) {
      float fb[16];
      bool okb[16];
#pragma unroll
      for (int j = 0; j < 16; ++j) {
        const int k = k0 + ((j >> 3) << 4) + (hi << 3) + (j & 7);
        okb[j] = colOK && (k < K);
        const unsigned off = okb[j] ? (unsigned)(k * N + col) : 0u;
        fb[j] = W[off];
      }
      v16h b;
#pragma unroll
      for (int j = 0; j < 16; ++j) b[j] = okb[j] ? (_Float16)fb[j] : (_Float16)0.f;
#pragma unroll
      for (int s = 0; s < 4; ++s) {
        const int row = rowBase + s * 16 + ml;
        const bool rOK = row < M;
        const unsigned rbase = rOK ? (unsigned)(row * K) : 0u;
        float fa[16];
        bool oka[16];
#pragma unroll
        for (int j = 0; j < 16; ++j) {
          const int k = k0 + ((j >> 3) << 4) + (hi << 3) + (j & 7);
          oka[j] = rOK && (k < K);
          const unsigned off = oka[j] ? (rbase + (unsigned)k) : 0u;
          fa[j] = A[off];
        }
        v16h a;
#pragma unroll
        for (int j = 0; j < 16; ++j) a[j] = oka[j] ? (_Float16)fa[j] : (_Float16)0.f;
        acc[s] = __builtin_amdgcn_wmma_f32_16x16x32_f16(false, a, false, b, (short)0, acc[s], false, false);
      }
    }
  }

  // -------- epilogue: D layout lane -> (n = ml, m = r + 8*hi) per subtile --------
  const float bv = bias ? bias[colOK ? col : 0] : 0.f;
  if (fullM && fullN) {
#pragma unroll
    for (int s = 0; s < 4; ++s) {
#pragma unroll
      for (int r = 0; r < 8; ++r) {
        const int m = rowBase + s * 16 + r + 8 * hi;
        float v = acc[s][r] + bv;
        if (ACT == 1) v = v > 0.f ? v : 0.f;
        else if (ACT == 2) v = sig_(v);
        else if (ACT == 3) v = softplus_(v);
        const long idx = (long)m * N + col;
        if (ACCUM) v += C[idx];
        C[idx] = v;
      }
    }
  } else {
#pragma unroll
    for (int s = 0; s < 4; ++s) {
#pragma unroll
      for (int r = 0; r < 8; ++r) {
        const int m = rowBase + s * 16 + r + 8 * hi;
        if (m < M && colOK) {
          float v = acc[s][r] + bv;
          if (ACT == 1) v = v > 0.f ? v : 0.f;
          else if (ACT == 2) v = sig_(v);
          else if (ACT == 3) v = softplus_(v);
          const long idx = (long)m * N + col;
          if (ACCUM) v += C[idx];
          C[idx] = v;
        }
      }
    }
  }
}

// ---------------- elementwise / reduction kernels ----------------
__global__ void k_fill(float* p, long n, float v){
  long i = blockIdx.x*(long)blockDim.x+threadIdx.x; if(i<n) p[i]=v;
}
__global__ void k_add(float* __restrict__ d, const float* __restrict__ s, long n){
  long i = blockIdx.x*(long)blockDim.x+threadIdx.x; if(i<n) d[i]+=s[i];
}
__global__ void k_transpose(const float* __restrict__ in, float* __restrict__ out, int R, int C){
  long i = blockIdx.x*(long)blockDim.x+threadIdx.x;
  if (i >= (long)R*C) return;
  int r = i / C, c = i % C;
  out[(long)c*R + r] = in[i];
}
__global__ void k_concat2(const float* __restrict__ A, const float* __restrict__ B,
                          float* __restrict__ out, int M, int D1, int D2){
  int D = D1 + D2;
  long i = blockIdx.x*(long)blockDim.x+threadIdx.x;
  if (i >= (long)M*D) return;
  int r = i / D, c = i % D;
  out[i] = (c < D1) ? A[(long)r*D1 + c] : B[(long)r*D2 + (c - D1)];
}
// per-row layernorm, one wave per row
__global__ __launch_bounds__(128)
void k_layernorm(const float* __restrict__ x, const float* __restrict__ g,
                 const float* __restrict__ b, float* __restrict__ y, int M, int D){
  int row = blockIdx.x * 4 + (threadIdx.x >> 5);
  if (row >= M) return;
  int lane = threadIdx.x & 31;
  const float* xr = x + (long)row * D;
  float s = 0.f, s2 = 0.f;
  for (int c = lane; c < D; c += 32) { float v = xr[c]; s += v; s2 += v*v; }
  for (int o = 16; o; o >>= 1) { s += __shfl_xor(s, o, 32); s2 += __shfl_xor(s2, o, 32); }
  float mean = s / D;
  float var = s2 / D - mean * mean;
  float inv = rsqrtf(var + 1e-5f);
  float* yr = y + (long)row * D;
  for (int c = lane; c < D; c += 32) yr[c] = g[c] * (xr[c] - mean) * inv + b[c];
}
__global__ void k_gru_combine(const float* __restrict__ gi, const float* __restrict__ gh,
                              const float* __restrict__ h, float* __restrict__ out,
                              int Nn, int Dh){
  long i = blockIdx.x*(long)blockDim.x+threadIdx.x;
  if (i >= (long)Nn*Dh) return;
  int n = i / Dh, c = i % Dh;
  const float* a = gi + (long)n*3*Dh;
  const float* b = gh + (long)n*3*Dh;
  float r = sig_(a[c] + b[c]);
  float z = sig_(a[Dh+c] + b[Dh+c]);
  float nn = tanhf(a[2*Dh+c] + r * b[2*Dh+c]);
  out[i] = (1.f - z) * nn + z * h[i];
}
// ---------------- edge gather / scatter ----------------
__global__ void k_static_cat(const float* __restrict__ ea, const float* __restrict__ xs_s,
                             const float* __restrict__ xs_d, const int* __restrict__ ei,
                             int E, float* __restrict__ out){
  long i = blockIdx.x*(long)blockDim.x+threadIdx.x;
  if (i >= (long)E*22) return;
  int e = i / 22, c = i % 22;
  float v;
  if (c < 6)       v = ea[(long)e*6 + c];
  else if (c < 14) v = xs_s[(long)ei[e]*8 + (c-6)];
  else             v = xs_d[(long)ei[E+e]*8 + (c-14)];
  out[i] = v;
}
__global__ void k_pair_cat(const float* __restrict__ hs, const float* __restrict__ hd,
                           const int* __restrict__ ei, int E, int D, float* __restrict__ out){
  long i = blockIdx.x*(long)blockDim.x+threadIdx.x;
  if (i >= (long)E*2*D) return;
  int e = i / (2*D), c = i % (2*D);
  out[i] = (c < D) ? hs[(long)ei[e]*D + c] : hd[(long)ei[E+e]*D + (c-D)];
}
__global__ void k_gather_rows(const float* __restrict__ src, const int* __restrict__ idx,
                              int E, int D, float* __restrict__ out){
  long i = blockIdx.x*(long)blockDim.x+threadIdx.x;
  if (i >= (long)E*D) return;
  int e = i / D, c = i % D;
  out[i] = src[(long)idx[e]*D + c];
}
__global__ void k_edge_scatter(const float* __restrict__ be, const float* __restrict__ g,
                               const float* __restrict__ v, const int* __restrict__ ei,
                               int E, float* __restrict__ msg){
  long i = blockIdx.x*(long)blockDim.x+threadIdx.x;
  if (i >= (long)E*MSG_) return;
  int e = i / MSG_, c = i % MSG_;
  int d = ei[E + e];
  atomicAdd(&msg[(long)d*MSG_ + c], be[e] * g[e] * v[i]);
}
// ---------------- GATv2 ----------------
__device__ __forceinline__ void atomicMaxF(float* addr, float val){
  if (val >= 0.f) atomicMax((int*)addr, __float_as_int(val));
  else            atomicMin((unsigned int*)addr, __float_as_uint(val));
}
__global__ void k_gat_scores(const float* __restrict__ xl, const float* __restrict__ xr,
                             const int* __restrict__ ei, const float* __restrict__ att,
                             int E, float* __restrict__ sc){
  long i = blockIdx.x*(long)blockDim.x+threadIdx.x;
  if (i >= (long)E*4) return;
  int e = i / 4, h = i % 4;
  int s = ei[e], d = ei[E+e];
  float acc = 0.f;
#pragma unroll
  for (int c = 0; c < 32; ++c) {
    float t = xl[(long)s*MSG_ + h*32 + c] + xr[(long)d*MSG_ + h*32 + c];
    t = t > 0.f ? t : 0.2f * t;
    acc += t * att[h*32 + c];
  }
  sc[i] = acc;
}
__global__ void k_gat_max(const float* __restrict__ sc, const int* __restrict__ ei,
                          int E, float* __restrict__ m){
  long i = blockIdx.x*(long)blockDim.x+threadIdx.x;
  if (i >= (long)E*4) return;
  int e = i / 4, h = i % 4;
  atomicMaxF(&m[(long)ei[E+e]*4 + h], sc[i]);
}
__global__ void k_gat_expsum(float* __restrict__ sc, const float* __restrict__ m,
                             const int* __restrict__ ei, int E, float* __restrict__ den){
  long i = blockIdx.x*(long)blockDim.x+threadIdx.x;
  if (i >= (long)E*4) return;
  int e = i / 4, h = i % 4;
  float v = __expf(sc[i] - m[(long)ei[E+e]*4 + h]);
  sc[i] = v;
  atomicAdd(&den[(long)ei[E+e]*4 + h], v);
}
__global__ void k_gat_apply(const float* __restrict__ ex, const float* __restrict__ den,
                            const float* __restrict__ xl, const int* __restrict__ ei,
                            int E, float* __restrict__ out){
  long i = blockIdx.x*(long)blockDim.x+threadIdx.x;
  if (i >= (long)E*MSG_) return;
  int e = i / MSG_, c = i % MSG_, h = c / 32;
  int s = ei[e], d = ei[E+e];
  float a = ex[(long)e*4 + h] / (den[(long)d*4 + h] + 1e-16f);
  atomicAdd(&out[(long)d*MSG_ + c], a * xl[(long)s*MSG_ + c]);
}
// ---------------- decoder ----------------
__global__ void k_dec_inp(const float* __restrict__ henc, const float* __restrict__ anchor,
                          const float* __restrict__ rain, int n0, int ch, int Ntot,
                          float* __restrict__ out){
  long i = blockIdx.x*(long)blockDim.x+threadIdx.x;
  long tot = (long)ch * T_FUT_ * 196;
  if (i >= tot) return;
  int row = i / 196, c = i % 196;
  int nl = row / T_FUT_, t = row % T_FUT_;
  int n = n0 + nl;
  float v;
  if (c < 192)      v = henc[(long)n*H_DIM_ + c];
  else if (c == 192) v = anchor[n];
  else if (c == 193) v = rain[(long)t*Ntot + n];
  else if (c == 194) v = sinf((float)t / 512.f);
  else               v = cosf((float)t / 512.f);
  out[i] = v;
}
// per-node 12x12 MHA, 8 heads x 32ch, qkv [ch*12, 768] -> out [ch*12, 256]
__global__ __launch_bounds__(128)
void k_dec_mha(const float* __restrict__ qkv, float* __restrict__ out, int ch){
  __shared__ float q[T_FUT_][DMOD_], k[T_FUT_][DMOD_], v[T_FUT_][DMOD_];
  int node = blockIdx.x;
  if (node >= ch) return;
  const float* base = qkv + (long)node * T_FUT_ * (3*DMOD_);
  for (int i = threadIdx.x; i < T_FUT_*DMOD_; i += blockDim.x) {
    int t = i / DMOD_, c = i % DMOD_;
    q[t][c] = base[t*(3*DMOD_) + c];
    k[t][c] = base[t*(3*DMOD_) + DMOD_ + c];
    v[t][c] = base[t*(3*DMOD_) + 2*DMOD_ + c];
  }
  __syncthreads();
  for (int p = threadIdx.x; p < NHEAD_*T_FUT_; p += blockDim.x) {
    int h = p / T_FUT_, t = p % T_FUT_;
    float sc[T_FUT_];
    float mx = -3e38f;
    for (int s = 0; s < T_FUT_; ++s) {
      float acc = 0.f;
#pragma unroll
      for (int c = 0; c < DH_; ++c) acc += q[t][h*DH_+c] * k[s][h*DH_+c];
      acc *= 0.17677669529663687f; // 1/sqrt(32)
      sc[s] = acc; mx = fmaxf(mx, acc);
    }
    float sum = 0.f;
    for (int s = 0; s < T_FUT_; ++s) { sc[s] = __expf(sc[s]-mx); sum += sc[s]; }
    float inv = 1.f / sum;
    float* orow = out + ((long)node*T_FUT_ + t)*DMOD_ + h*DH_;
#pragma unroll
    for (int c = 0; c < DH_; ++c) {
      float o = 0.f;
      for (int s = 0; s < T_FUT_; ++s) o += sc[s] * v[s][h*DH_+c];
      orow[c] = o * inv;
    }
  }
}
__global__ void k_dec_write(const float* __restrict__ delta, const float* __restrict__ anchor,
                            int n0, int ch, int Ntot, float* __restrict__ out){
  long i = blockIdx.x*(long)blockDim.x+threadIdx.x;
  if (i >= (long)ch*T_FUT_) return;
  int nl = i / T_FUT_, t = i % T_FUT_;
  out[(long)t*Ntot + n0 + nl] = delta[i] + anchor[n0 + nl];
}

// ---------------- parameter offsets (jax pytree: dicts flattened by sorted key) ----------------
struct Lin { long b, w; };
struct Ln  { long b, g; };
struct Mlp { Lin l1, l2; };
struct Sd  { Lin base; Mlp emb; Mlp gate; Mlp payload; };
struct Gat { long att, bias; Mlp ffn; long wl, wr, wres; };
struct Layer { Lin ff1, ff2; Ln ln1, ln2; long out_b, out_w, qkv_b, qkv_w; };
struct Dec { Lin inp; Layer layers[2]; Lin out; };
struct Node { Ln dyn_norm; Lin dyn_proj; long bhh, bih, whh, wih; Ln h_norm, msg_norm; };
struct Params { Dec dec1, dec2; Sd mp11; Gat mp12; Sd mp22; Node nd1, nd2; };

static Params build_params() {
  long cur = 0;
  auto take = [&](long n){ long o = cur; cur += n; return o; };
  auto lin  = [&](int din,int dout){ Lin p; p.b=take(dout); p.w=take((long)din*dout); return p; };
  auto ln   = [&](int d){ Ln p; p.b=take(d); p.g=take(d); return p; };
  auto mlp  = [&](int din,int hid,int dout){ Mlp p; p.l1=lin(din,hid); p.l2=lin(hid,dout); return p; };
  auto layer= [&](){ Layer L; L.ff1=lin(DMOD_,FFN_); L.ff2=lin(FFN_,DMOD_);
                     L.ln1=ln(DMOD_); L.ln2=ln(DMOD_);
                     L.out_b=take(DMOD_); L.out_w=take((long)DMOD_*DMOD_);
                     L.qkv_b=take(3*DMOD_); L.qkv_w=take((long)DMOD_*3*DMOD_); return L; };
  auto dec  = [&](){ Dec d; d.inp=lin(H_DIM_+4,DMOD_); d.layers[0]=layer(); d.layers[1]=layer();
                     d.out=lin(DMOD_,1); return d; };
  auto sd   = [&](){ Sd s; s.base=lin(HID_,1); s.emb=mlp(22,HID_,HID_);
                     s.gate=mlp(2*H_DIM_,HID_,1); s.payload=mlp(H_DIM_,HID_,MSG_); return s; };
  auto node = [&](){ Node n; n.dyn_norm=ln(MSG_); n.dyn_proj=lin(4,MSG_);
                     n.bhh=take(3*H_DIM_); n.bih=take(3*H_DIM_);
                     n.whh=take((long)3*H_DIM_*H_DIM_); n.wih=take((long)3*H_DIM_*2*MSG_);
                     n.h_norm=ln(H_DIM_); n.msg_norm=ln(MSG_); return n; };
  Params p;
  p.dec1 = dec(); p.dec2 = dec();          // 'dec' (oneD, twoD)
  p.mp11 = sd();                           // 'mp11'
  Gat g; g.att=take(128); g.bias=take(128); g.ffn=mlp(MSG_,HID_,MSG_);
  g.wl=take((long)H_DIM_*MSG_); g.wr=take((long)H_DIM_*MSG_); g.wres=take((long)H_DIM_*MSG_);
  p.mp12 = g;                              // 'mp12'
  p.mp22 = sd();                           // 'mp22'
  p.nd1 = node(); p.nd2 = node();          // 'node' (oneD, twoD)
  return p;
}

// ---------------- host-side launch helpers ----------------
static void gemm(hipStream_t s, int act, int accum,
                 const float* A, const float* W, const float* bias,
                 float* C, int M, int K, int N) {
  int tiles = CDIV(M,64) * CDIV(N,16);
  dim3 g(CDIV(tiles,4)), b(128);
  if (accum) { gemm_k<0,1><<<g,b,0,s>>>(A,W,bias,C,M,K,N); return; }
  switch (act) {
    case 1:  gemm_k<1,0><<<g,b,0,s>>>(A,W,bias,C,M,K,N); break;
    case 2:  gemm_k<2,0><<<g,b,0,s>>>(A,W,bias,C,M,K,N); break;
    case 3:  gemm_k<3,0><<<g,b,0,s>>>(A,W,bias,C,M,K,N); break;
    default: gemm_k<0,0><<<g,b,0,s>>>(A,W,bias,C,M,K,N); break;
  }
}

extern "C" void kernel_launch(void* const* d_in, const int* in_sizes, int n_in,
                              void* d_out, int out_size, void* d_ws, size_t ws_size,
                              hipStream_t stream) {
  (void)in_sizes; (void)n_in; (void)out_size; (void)ws_size;
  const float* h0_1  = (const float*)d_in[0];
  const float* h0_2  = (const float*)d_in[1];
  const float* xs1   = (const float*)d_in[2];
  const float* xs2   = (const float*)d_in[3];
  const float* xd1   = (const float*)d_in[4];
  const float* xd2   = (const float*)d_in[5];
  const float* ea11  = (const float*)d_in[6];
  const float* ea22  = (const float*)d_in[7];
  const float* rain1 = (const float*)d_in[8];
  const float* rain2 = (const float*)d_in[9];
  const float* anc1  = (const float*)d_in[10];
  const float* anc2  = (const float*)d_in[11];
  const float* P     = (const float*)d_in[12];
  const int*   ei11  = (const int*)d_in[13];
  const int*   ei22  = (const int*)d_in[14];
  const int*   ei12  = (const int*)d_in[15];
  float* out = (float*)d_out;

  const Params pp = build_params();

  // workspace bump allocator
  char* wsb = (char*)d_ws;
  size_t wo = 0;
  auto wal = [&](size_t floats)->float* {
    wo = (wo + 255) & ~(size_t)255;
    float* p = (float*)(wsb + wo);
    wo += floats * sizeof(float);
    return p;
  };
  float* wihT1 = wal((long)256*576); float* whhT1 = wal((long)192*576);
  float* wihT2 = wal((long)256*576); float* whhT2 = wal((long)192*576);
  float* b11 = wal(E11_); float* b22 = wal(E22_);
  float* h1a = wal((long)N1_*H_DIM_); float* h1b = wal((long)N1_*H_DIM_);
  float* h2a = wal((long)N2_*H_DIM_); float* h2b = wal((long)N2_*H_DIM_);
  float* hidE    = wal((long)E22_*HID_);
  float* gateInE = wal((long)E22_*2*H_DIM_);  // also aliases embE in static phase
  float* embE    = gateInE;
  float* payInE  = wal((long)E22_*H_DIM_);    // also aliases catE in static phase
  float* catE    = payInE;
  float* gateE   = wal(E22_);
  float* payE    = wal((long)E22_*MSG_);
  float* msg1 = wal((long)N1_*MSG_); float* msg2 = wal((long)N2_*MSG_);
  float* xl = wal((long)N1_*MSG_);   float* xr = wal((long)N2_*MSG_);
  float* esc = wal((long)E12_*4); float* m2 = wal((long)N2_*4); float* den2 = wal((long)N2_*4);
  float* gatOut = wal((long)N2_*MSG_); float* gatHid = wal((long)N2_*HID_); float* gatFin = wal((long)N2_*MSG_);
  float* dynB = wal((long)N2_*MSG_); float* meB = wal((long)N2_*MSG_);
  float* catGru = wal((long)N2_*2*MSG_);
  float* gi = wal((long)N2_*3*H_DIM_); float* gh = wal((long)N2_*3*H_DIM_);
  float* hg = wal((long)N2_*H_DIM_);
  const long TOK = (long)DCH_ * T_FUT_;
  float* inpB  = wal(TOK*(H_DIM_+4));
  float* xB    = wal(TOK*DMOD_);
  float* lnB   = wal(TOK*DMOD_);
  float* qkvB  = wal(TOK*3*DMOD_);
  float* attnB = wal(TOK*DMOD_);
  float* ffB   = wal(TOK*FFN_);
  float* dltB  = wal(TOK);

  auto fill = [&](float* p, long n, float v){ k_fill<<<CDIV(n,256),256,0,stream>>>(p,n,v); };
  auto LN = [&](const float* x, const Ln& l, float* y, int M, int D){
    k_layernorm<<<CDIV(M,4),128,0,stream>>>(x, P+l.g, P+l.b, y, M, D);
  };

  // ---- hoisted: GRU weight transposes (wih [576,din] -> [din,576]) ----
  { long n=(long)576*256; k_transpose<<<CDIV(n,256),256,0,stream>>>(P+pp.nd1.wih, wihT1, 576, 256); }
  { long n=(long)576*192; k_transpose<<<CDIV(n,256),256,0,stream>>>(P+pp.nd1.whh, whhT1, 576, 192); }
  { long n=(long)576*256; k_transpose<<<CDIV(n,256),256,0,stream>>>(P+pp.nd2.wih, wihT2, 576, 256); }
  { long n=(long)576*192; k_transpose<<<CDIV(n,256),256,0,stream>>>(P+pp.nd2.whh, whhT2, 576, 192); }

  // ---- hoisted: static edge base weights b11 / b22 ----
  auto sd_static = [&](const Sd& s, const float* ea, const float* xs, const int* ei,
                       int E, float* bout){
    long n = (long)E*22;
    k_static_cat<<<CDIV(n,256),256,0,stream>>>(ea, xs, xs, ei, E, catE);
    gemm(stream, 1, 0, catE, P+s.emb.l1.w, P+s.emb.l1.b, hidE, E, 22, HID_);
    gemm(stream, 0, 0, hidE, P+s.emb.l2.w, P+s.emb.l2.b, embE, E, HID_, HID_);
    gemm(stream, 3, 0, embE, P+s.base.w,  P+s.base.b,  bout, E, HID_, 1);
  };
  sd_static(pp.mp11, ea11, xs1, ei11, E11_, b11);
  sd_static(pp.mp22, ea22, xs2, ei22, E22_, b22);

  // ---- scan over T_HIST ----
  float *h1 = h1a, *h1n = h1b, *h2 = h2a, *h2n = h2b;
  hipMemcpyAsync(h1, h0_1, (long)N1_*H_DIM_*sizeof(float), hipMemcpyDeviceToDevice, stream);
  hipMemcpyAsync(h2, h0_2, (long)N2_*H_DIM_*sizeof(float), hipMemcpyDeviceToDevice, stream);

  auto sd_dyn = [&](const Sd& s, const float* be, const float* h, const int* ei,
                    int E, int Nd, float* msg){
    fill(msg, (long)Nd*MSG_, 0.f);
    { long n=(long)E*2*H_DIM_; k_pair_cat<<<CDIV(n,256),256,0,stream>>>(h, h, ei, E, H_DIM_, gateInE); }
    gemm(stream, 1, 0, gateInE, P+s.gate.l1.w, P+s.gate.l1.b, hidE, E, 2*H_DIM_, HID_);
    gemm(stream, 2, 0, hidE,    P+s.gate.l2.w, P+s.gate.l2.b, gateE, E, HID_, 1);
    { long n=(long)E*H_DIM_; k_gather_rows<<<CDIV(n,256),256,0,stream>>>(h, ei, E, H_DIM_, payInE); }
    gemm(stream, 1, 0, payInE, P+s.payload.l1.w, P+s.payload.l1.b, hidE, E, H_DIM_, HID_);
    gemm(stream, 0, 0, hidE,   P+s.payload.l2.w, P+s.payload.l2.b, payE, E, HID_, MSG_);
    { long n=(long)E*MSG_; k_edge_scatter<<<CDIV(n,256),256,0,stream>>>(be, gateE, payE, ei, E, msg); }
  };
  auto upd = [&](const Node& nd, const float* wihT, const float* whhT,
                 const float* h, const float* msg, const float* xdt, int Nn, float* hnew){
    gemm(stream, 0, 0, xdt, P+nd.dyn_proj.w, P+nd.dyn_proj.b, dynB, Nn, 4, MSG_);
    LN(dynB, nd.dyn_norm, dynB, Nn, MSG_);
    LN(msg,  nd.msg_norm, meB,  Nn, MSG_);
    { long n=(long)Nn*2*MSG_; k_concat2<<<CDIV(n,256),256,0,stream>>>(dynB, meB, catGru, Nn, MSG_, MSG_); }
    gemm(stream, 0, 0, catGru, wihT, P+nd.bih, gi, Nn, 2*MSG_, 3*H_DIM_);
    gemm(stream, 0, 0, h,      whhT, P+nd.bhh, gh, Nn, H_DIM_, 3*H_DIM_);
    { long n=(long)Nn*H_DIM_; k_gru_combine<<<CDIV(n,256),256,0,stream>>>(gi, gh, h, hg, Nn, H_DIM_); }
    LN(hg, nd.h_norm, hnew, Nn, H_DIM_);
  };

  for (int t = 0; t < T_HIST_; ++t) {
    sd_dyn(pp.mp11, b11, h1, ei11, E11_, N1_, msg1);
    sd_dyn(pp.mp22, b22, h2, ei22, E22_, N2_, msg2);
    // GATv2 1D -> 2D
    gemm(stream, 0, 0, h1, P+pp.mp12.wl, nullptr, xl, N1_, H_DIM_, MSG_);
    gemm(stream, 0, 0, h2, P+pp.mp12.wr, nullptr, xr, N2_, H_DIM_, MSG_);
    { long n=(long)E12_*4; k_gat_scores<<<CDIV(n,256),256,0,stream>>>(xl, xr, ei12, P+pp.mp12.att, E12_, esc); }
    fill(m2, (long)N2_*4, -3e38f);
    { long n=(long)E12_*4; k_gat_max<<<CDIV(n,256),256,0,stream>>>(esc, ei12, E12_, m2); }
    fill(den2, (long)N2_*4, 0.f);
    { long n=(long)E12_*4; k_gat_expsum<<<CDIV(n,256),256,0,stream>>>(esc, m2, ei12, E12_, den2); }
    fill(gatOut, (long)N2_*MSG_, 0.f);
    { long n=(long)E12_*MSG_; k_gat_apply<<<CDIV(n,256),256,0,stream>>>(esc, den2, xl, ei12, E12_, gatOut); }
    gemm(stream, 0, 1, h2, P+pp.mp12.wres, P+pp.mp12.bias, gatOut, N2_, H_DIM_, MSG_); // += residual + bias
    gemm(stream, 1, 0, gatOut, P+pp.mp12.ffn.l1.w, P+pp.mp12.ffn.l1.b, gatHid, N2_, MSG_, HID_);
    gemm(stream, 0, 0, gatHid, P+pp.mp12.ffn.l2.w, P+pp.mp12.ffn.l2.b, gatFin, N2_, HID_, MSG_);
    { long n=(long)N2_*MSG_; k_add<<<CDIV(n,256),256,0,stream>>>(msg2, gatFin, n); }
    // GRU node updates
    upd(pp.nd1, wihT1, whhT1, h1, msg1, xd1 + (long)t*N1_*4, N1_, h1n);
    upd(pp.nd2, wihT2, whhT2, h2, msg2, xd2 + (long)t*N2_*4, N2_, h2n);
    float* tp;
    tp = h1; h1 = h1n; h1n = tp;
    tp = h2; h2 = h2n; h2n = tp;
  }

  // ---- decoders (chunked over nodes) ----
  auto run_decoder = [&](const Dec& D, const float* henc, const float* anchor,
                         const float* rain, int Nn, float* outBase){
    for (int n0 = 0; n0 < Nn; n0 += DCH_) {
      int ch = (Nn - n0 < DCH_) ? (Nn - n0) : DCH_;
      int tok = ch * T_FUT_;
      { long n=(long)tok*(H_DIM_+4);
        k_dec_inp<<<CDIV(n,256),256,0,stream>>>(henc, anchor, rain, n0, ch, Nn, inpB); }
      gemm(stream, 0, 0, inpB, P+D.inp.w, P+D.inp.b, xB, tok, H_DIM_+4, DMOD_);
      for (int l = 0; l < 2; ++l) {
        const Layer& L = D.layers[l];
        LN(xB, L.ln1, lnB, tok, DMOD_);
        gemm(stream, 0, 0, lnB, P+L.qkv_w, P+L.qkv_b, qkvB, tok, DMOD_, 3*DMOD_);
        k_dec_mha<<<ch,128,0,stream>>>(qkvB, attnB, ch);
        gemm(stream, 0, 1, attnB, P+L.out_w, P+L.out_b, xB, tok, DMOD_, DMOD_);  // x += proj
        LN(xB, L.ln2, lnB, tok, DMOD_);
        gemm(stream, 1, 0, lnB, P+L.ff1.w, P+L.ff1.b, ffB, tok, DMOD_, FFN_);
        gemm(stream, 0, 1, ffB, P+L.ff2.w, P+L.ff2.b, xB, tok, FFN_, DMOD_);     // x += ffn
      }
      gemm(stream, 0, 0, xB, P+D.out.w, P+D.out.b, dltB, tok, DMOD_, 1);
      { long n=tok; k_dec_write<<<CDIV(n,256),256,0,stream>>>(dltB, anchor, n0, ch, Nn, outBase); }
    }
  };
  run_decoder(pp.dec1, h1, anc1, rain1, N1_, out);
  run_decoder(pp.dec2, h2, anc2, rain2, N2_, out + (long)T_FUT_*N1_);
}
